// GNN_49392123904049
// MI455X (gfx1250) — compile-verified, matrix operand order checked by reading, not run on hardware
//
#include <hip/hip_runtime.h>

// ---------------------------------------------------------------------------
// GNN (jraph GraphNetwork) for MI455X / gfx1250.
// Dense math: v_wmma_f32_16x16x32_f16 (f16 in, f32 acc). Activations kept in
// f16 globally (halves gather bandwidth); gathers done with
// GLOBAL_LOAD_ASYNC_TO_LDS_B128 (ASYNCcnt) for the f16 row segments;
// segment-sums via fp32 atomics fused into the producing kernels.
// ---------------------------------------------------------------------------

#define N_NODES  50000
#define N_EDGES  400000
#define N_GRAPHS 256
#define D_NODE   9
#define D_EDGE   3
#define LATENT   64
#define STEPS    3

typedef _Float16 half_t;
typedef __attribute__((ext_vector_type(16))) half_t v16h;
typedef __attribute__((ext_vector_type(8)))  half_t v8h;
typedef __attribute__((ext_vector_type(8)))  float  v8f;

// Fragment coordinate math (16-bit A/B fragments, wave32):
//   K(l,h) = (h&7) + ((h>>3)<<4) + ((l>>4)<<3)      (A: row=l&15, B: col=l&15)
// For row-major A data this means per lane the 16 halves are two contiguous
// 8-half runs at K = base+((l>>4)<<3) and +16.

// Async gather: one 16-byte global->LDS copy, tracked on ASYNCcnt.
__device__ __forceinline__ void async_ld_b128(unsigned ldsOff, const void* g) {
    asm volatile("global_load_async_to_lds_b128 %0, %1, off"
                 :: "v"(ldsOff), "v"(g) : "memory");
}
__device__ __forceinline__ void wait_async0() {
    asm volatile("s_wait_asynccnt 0x0" ::: "memory");
}
__device__ __forceinline__ unsigned lds_off(const void* p) {
    return (unsigned)(size_t)p;     // generic LDS ptr: low 32 bits = LDS offset
}

// --------------------------- weight packing --------------------------------
// Pack W[din][dout] (f32, row-major) into f16 B-fragment tiles:
// layout [kt][nt][lane(32)][16 halves], tile = 16 cols x 32 k.
__global__ void gnn_pack_w(const float* __restrict__ W, half_t* __restrict__ Wp,
                           int din, int dout) {
    int total = din * dout;
    int dout_t = dout >> 4;
    for (int idx = blockIdx.x * blockDim.x + threadIdx.x; idx < total;
         idx += gridDim.x * blockDim.x) {
        int h    = idx & 15;
        int lane = (idx >> 4) & 31;
        int tile = idx >> 9;                 // kt*dout_t + nt
        int nt   = tile % dout_t;
        int kt   = tile / dout_t;
        int k = kt * 32 + ((h & 7) | ((h >> 3) << 4) | ((lane >> 4) << 3));
        int n = nt * 16 + (lane & 15);
        Wp[idx] = (half_t)W[k * dout + n];
    }
}

__global__ void gnn_zero_f32(float* __restrict__ p, long n) {
    long i = (long)blockIdx.x * blockDim.x + threadIdx.x;
    long stride = (long)gridDim.x * blockDim.x;
    for (; i < n; i += stride) p[i] = 0.f;
}

// --------------------------- embedding -------------------------------------
__global__ void gnn_embed(const float* __restrict__ X, const float* __restrict__ W,
                          const float* __restrict__ b, half_t* __restrict__ out,
                          int rows, int din) {
    long idx = (long)blockIdx.x * blockDim.x + threadIdx.x;
    long total = (long)rows * LATENT;
    if (idx >= total) return;
    int  c = (int)(idx & (LATENT - 1));
    long r = idx >> 6;
    float acc = b[c];
    for (int k = 0; k < din; ++k) acc += X[r * din + k] * W[k * LATENT + c];
    out[idx] = (half_t)acc;
}

// --------------------------- fused MLP block --------------------------------
// 64 rows per block, 256 threads (8 waves). din=224, hid=128, out=64.
// MODE 0 = edge update (gather edges/nodes[s]/nodes[r]/g; scatter sent/recv/e_agg)
// MODE 1 = node update (gather nodes/sent/recv/g; scatter n_agg)
template <int MODE>
__global__ __launch_bounds__(256) void gnn_mlp(
    const half_t* __restrict__ featA,    // 64-wide f16 (edges or nodes)
    const half_t* __restrict__ featB,    // nodes f16 (MODE 0) / unused
    const float*  __restrict__ sentIn,   // MODE 1: sent agg
    const float*  __restrict__ recvIn,   // MODE 1: recv agg
    const float*  __restrict__ gIn,      // 256x32 f32
    const int*    __restrict__ senders,
    const int*    __restrict__ receivers,
    const int*    __restrict__ gids,     // per-row graph id
    const half_t* __restrict__ W1p, const float* __restrict__ b1,
    const float*  __restrict__ lnS, const float* __restrict__ lnB,
    const half_t* __restrict__ W2p, const float* __restrict__ b2,
    half_t* __restrict__ outF16,         // rows x 64
    float*  __restrict__ acc0,           // MODE0: sent   MODE1: n_agg
    float*  __restrict__ acc1,           // MODE0: recv
    float*  __restrict__ acc2,           // MODE0: e_agg
    int nRows) {
    constexpr int XS = 232;                        // row stride in halves (bank skew)
    // Xrow: 64 x 232 halves row-major (29.0 KB); later reused as Hpack (16 KB).
    __shared__ __align__(32) half_t Xrow[64 * XS];
    __shared__ float Hf32[64 * 128];               // 32 KB
    __shared__ float rMean[64], rRstd[64];
    half_t* Hpack = Xrow;

    const int tid  = threadIdx.x;
    const int wave = tid >> 5;
    const int lane = tid & 31;
    const int ROW0 = blockIdx.x * 64;

    // ---- zero-fill rows past nRows (boundary blocks only; uniform branch) ----
    if (ROW0 + 64 > nRows) {
        for (int idx = tid; idx < 64 * XS; idx += 256) {
            int m = idx / XS;
            if (ROW0 + m >= nRows) Xrow[idx] = (half_t)0.f;
        }
    }

    // ---- gather: f16 row segments via async global->LDS b128 copies ----
    // self features: 64 rows x 64 halves (contiguous in global)
    for (int idx = tid; idx < 512; idx += 256) {
        int m = idx >> 3, c = idx & 7;
        int row = ROW0 + m;
        if (row < nRows)
            async_ld_b128(lds_off(&Xrow[m * XS + c * 8]),
                          featA + (long)row * 64 + c * 8);
    }
    if (MODE == 0) {
        // sender / receiver node rows: 2 x 64 rows x 64 halves
        for (int idx = tid; idx < 1024; idx += 256) {
            int m = idx >> 4, seg = (idx >> 3) & 1, c = idx & 7;
            int row = ROW0 + m;
            if (row < nRows) {
                int nid = seg ? receivers[row] : senders[row];
                async_ld_b128(lds_off(&Xrow[m * XS + 64 + seg * 64 + c * 8]),
                              featB + (long)nid * 64 + c * 8);
            }
        }
    } else {
        // sent/recv aggregates are fp32 (atomic targets): load + convert
        for (int idx = tid; idx < 64 * 128; idx += 256) {
            int m = idx >> 7, c = idx & 127;
            int row = ROW0 + m;
            if (row < nRows) {
                float v = (c < 64) ? sentIn[(long)row * 64 + c]
                                   : recvIn[(long)row * 64 + (c - 64)];
                Xrow[m * XS + 64 + c] = (half_t)v;
            }
        }
    }
    // global-feature columns (fp32 -> f16)
    for (int idx = tid; idx < 64 * 32; idx += 256) {
        int m = idx >> 5, c = idx & 31;
        int row = ROW0 + m;
        if (row < nRows)
            Xrow[m * XS + 192 + c] = (half_t)gIn[(long)gids[row] * 32 + c];
    }
    wait_async0();
    __syncthreads();

    // ---- layer 1: (64x224) @ (224x128) -> Hf32, 32 tiles over 8 waves ----
    for (int t = wave * 4; t < wave * 4 + 4; ++t) {
        int mt = t >> 3, nt = t & 7;
        v8f acc = {};
        const half_t* rowp =
            &Xrow[(mt * 16 + (lane & 15)) * XS + ((lane >> 4) << 3)];
#pragma unroll
        for (int kt = 0; kt < 7; ++kt) {
            v8h lo = *(const v8h*)(rowp + kt * 32);
            v8h hi = *(const v8h*)(rowp + kt * 32 + 16);
            v16h a = __builtin_shufflevector(lo, hi, 0, 1, 2, 3, 4, 5, 6, 7,
                                             8, 9, 10, 11, 12, 13, 14, 15);
            v16h bf = *(const v16h*)&W1p[((kt * 8 + nt) * 32 + lane) << 4];
            acc = __builtin_amdgcn_wmma_f32_16x16x32_f16(
                false, a, false, bf, (short)0, acc, false, false);
        }
        int col = nt * 16 + (lane & 15);
        int rowBase = mt * 16 + ((lane >> 4) << 3);
        float bias = b1[col];
#pragma unroll
        for (int r = 0; r < 8; ++r)
            Hf32[(rowBase + r) * 128 + col] = acc[r] + bias;
    }
    __syncthreads();

    // ---- LayerNorm stats per row ----
    if (tid < 64) {
        float s = 0.f, s2 = 0.f;
        for (int c = 0; c < 128; ++c) {
            float v = Hf32[tid * 128 + c];
            s += v; s2 += v * v;
        }
        float m = s * (1.f / 128.f);
        float var = s2 * (1.f / 128.f) - m * m;
        rMean[tid] = m;
        rRstd[tid] = rsqrtf(var + 1e-6f);
    }
    __syncthreads();

    // ---- normalize + scale/bias + ReLU -> Hpack (A-fragment layout) ----
    for (int idx = tid; idx < 64 * 128; idx += 256) {
        int m = idx >> 7, c = idx & 127;
        float v = (Hf32[idx] - rMean[m]) * rRstd[m] * lnS[c] + lnB[c];
        v = fmaxf(v, 0.f);
        int kt = c >> 5, kk = c & 31;
        int h = ((kk >> 4) << 3) | (kk & 7);
        int l = (m & 15) | ((kk & 8) << 1);
        Hpack[(((((m >> 4) * 4) + kt) * 32 + l) << 4) + h] = (half_t)v;
    }
    __syncthreads();

    // ---- layer 2: (64x128) @ (128x64) -> out + fused scatter atomics ----
    for (int t = wave * 2; t < wave * 2 + 2; ++t) {
        int mt = t >> 2, nt = t & 3;
        v8f acc = {};
#pragma unroll
        for (int kt = 0; kt < 4; ++kt) {
            v16h a = *(const v16h*)&Hpack[((mt * 4 + kt) * 32 + lane) << 4];
            v16h bf = *(const v16h*)&W2p[((kt * 4 + nt) * 32 + lane) << 4];
            acc = __builtin_amdgcn_wmma_f32_16x16x32_f16(
                false, a, false, bf, (short)0, acc, false, false);
        }
        int col = nt * 16 + (lane & 15);
        int rowBase = mt * 16 + ((lane >> 4) << 3);
        float bias = b2[col];
#pragma unroll
        for (int r = 0; r < 8; ++r) {
            int row = ROW0 + rowBase + r;
            if (row < nRows) {
                float v = acc[r] + bias;
                outF16[(long)row * 64 + col] = (half_t)v;
                if (MODE == 0) {
                    unsafeAtomicAdd(&acc0[(long)senders[row] * 64 + col], v);
                    unsafeAtomicAdd(&acc1[(long)receivers[row] * 64 + col], v);
                    unsafeAtomicAdd(&acc2[(long)gids[row] * 64 + col], v);
                } else {
                    unsafeAtomicAdd(&acc0[(long)gids[row] * 64 + col], v);
                }
            }
        }
    }
}

// --------------------------- global MLP -------------------------------------
// One block per graph; 160 -> 128(LN,ReLU) -> 64(LN,ReLU) -> 32. Tiny; VALU.
__global__ __launch_bounds__(128) void gnn_global_mlp(
    const float* __restrict__ nAgg, const float* __restrict__ eAgg,
    const float* __restrict__ gIn,
    const float* __restrict__ W1, const float* __restrict__ B1,
    const float* __restrict__ S1, const float* __restrict__ Bi1,
    const float* __restrict__ W2, const float* __restrict__ B2,
    const float* __restrict__ S2, const float* __restrict__ Bi2,
    const float* __restrict__ W3, const float* __restrict__ B3,
    float* __restrict__ gOut) {
    __shared__ float xin[160];
    __shared__ float h[128];
    __shared__ float stat[2];
    int g = blockIdx.x, tid = threadIdx.x;

    if (tid < 64) {
        xin[tid]      = nAgg[g * 64 + tid];
        xin[64 + tid] = eAgg[g * 64 + tid];
    }
    if (tid < 32) xin[128 + tid] = gIn[g * 32 + tid];
    __syncthreads();

    float a = B1[tid];
    for (int k = 0; k < 160; ++k) a += xin[k] * W1[k * 128 + tid];
    h[tid] = a;
    __syncthreads();
    if (tid == 0) {
        float s = 0.f, s2 = 0.f;
        for (int c = 0; c < 128; ++c) { float v = h[c]; s += v; s2 += v * v; }
        float m = s / 128.f;
        stat[0] = m; stat[1] = rsqrtf(s2 / 128.f - m * m + 1e-6f);
    }
    __syncthreads();
    float v1 = fmaxf((h[tid] - stat[0]) * stat[1] * S1[tid] + Bi1[tid], 0.f);
    __syncthreads();
    h[tid] = v1;
    __syncthreads();

    float a2 = 0.f;
    if (tid < 64) {
        a2 = B2[tid];
        for (int k = 0; k < 128; ++k) a2 += h[k] * W2[k * 64 + tid];
    }
    __syncthreads();
    if (tid < 64) h[tid] = a2;
    __syncthreads();
    if (tid == 0) {
        float s = 0.f, s2 = 0.f;
        for (int c = 0; c < 64; ++c) { float v = h[c]; s += v; s2 += v * v; }
        float m = s / 64.f;
        stat[0] = m; stat[1] = rsqrtf(s2 / 64.f - m * m + 1e-6f);
    }
    __syncthreads();
    float v2 = 0.f;
    if (tid < 64) v2 = fmaxf((h[tid] - stat[0]) * stat[1] * S2[tid] + Bi2[tid], 0.f);
    __syncthreads();
    if (tid < 64) h[tid] = v2;
    __syncthreads();
    if (tid < 32) {
        float a3 = B3[tid];
        for (int k = 0; k < 64; ++k) a3 += h[k] * W3[k * 32 + tid];
        gOut[g * 32 + tid] = a3;
    }
}

// --------------------------- host orchestration -----------------------------
// Input leaves, dict-insertion order:
//  0 node_feats  1 edge_feats
//  2 embed_node.W  3 .b   4 embed_edge.W  5 .b
//  per step s (base = 6 + 22*s):
//    edge:   +0 W1 +1 b1 +2 scale +3 bias +4 W2 +5 b2
//    node:   +6..+11 (same pattern)
//    global: +12 W1 +13 b1 +14 s1 +15 bi1 +16 W2 +17 b2 +18 s2 +19 bi2 +20 W3 +21 b3
//  72 senders  73 receivers  74 node_graph_ids  75 edge_graph_ids
extern "C" void kernel_launch(void* const* d_in, const int* in_sizes, int n_in,
                              void* d_out, int out_size, void* d_ws, size_t ws_size,
                              hipStream_t stream) {
    const float* node_feats = (const float*)d_in[0];
    const float* edge_feats = (const float*)d_in[1];
    const float* enW = (const float*)d_in[2];
    const float* enB = (const float*)d_in[3];
    const float* eeW = (const float*)d_in[4];
    const float* eeB = (const float*)d_in[5];
    const int* senders   = (const int*)d_in[72];
    const int* receivers = (const int*)d_in[73];
    const int* node_gid  = (const int*)d_in[74];
    const int* edge_gid  = (const int*)d_in[75];

    // ---- carve workspace ----
    char* wp = (char*)d_ws;
    auto carve = [&](size_t bytes) {
        char* p = wp;
        wp += (bytes + 255) & ~(size_t)255;
        return (void*)p;
    };
    half_t* edgesH[2] = {(half_t*)carve((size_t)N_EDGES * 64 * 2),
                         (half_t*)carve((size_t)N_EDGES * 64 * 2)};
    half_t* nodesH[2] = {(half_t*)carve((size_t)N_NODES * 64 * 2),
                         (half_t*)carve((size_t)N_NODES * 64 * 2)};
    float* sent  = (float*)carve((size_t)N_NODES * 64 * 4);   // sent+recv contiguous
    float* recv  = (float*)carve((size_t)N_NODES * 64 * 4);
    float* nAgg  = (float*)carve((size_t)N_GRAPHS * 64 * 4);  // nAgg+eAgg contiguous
    float* eAgg  = (float*)carve((size_t)N_GRAPHS * 64 * 4);
    float* gBuf[2] = {(float*)carve((size_t)N_GRAPHS * 32 * 4),
                      (float*)carve((size_t)N_GRAPHS * 32 * 4)};
    half_t* eW1p[STEPS]; half_t* eW2p[STEPS];
    half_t* nW1p[STEPS]; half_t* nW2p[STEPS];
    for (int s = 0; s < STEPS; ++s) {
        eW1p[s] = (half_t*)carve((size_t)224 * 128 * 2);
        eW2p[s] = (half_t*)carve((size_t)128 * 64 * 2);
        nW1p[s] = (half_t*)carve((size_t)224 * 128 * 2);
        nW2p[s] = (half_t*)carve((size_t)128 * 64 * 2);
    }

    // ---- pack weights to f16 fragment layout ----
    for (int s = 0; s < STEPS; ++s) {
        int base = 6 + 22 * s;
        gnn_pack_w<<<(224 * 128 + 255) / 256, 256, 0, stream>>>(
            (const float*)d_in[base + 0], eW1p[s], 224, 128);
        gnn_pack_w<<<(128 * 64 + 255) / 256, 256, 0, stream>>>(
            (const float*)d_in[base + 4], eW2p[s], 128, 64);
        gnn_pack_w<<<(224 * 128 + 255) / 256, 256, 0, stream>>>(
            (const float*)d_in[base + 6], nW1p[s], 224, 128);
        gnn_pack_w<<<(128 * 64 + 255) / 256, 256, 0, stream>>>(
            (const float*)d_in[base + 10], nW2p[s], 128, 64);
    }

    // ---- init g = 0 ----
    gnn_zero_f32<<<(N_GRAPHS * 32 + 255) / 256, 256, 0, stream>>>(
        gBuf[0], (long)N_GRAPHS * 32);

    // ---- embeddings ----
    gnn_embed<<<((long)N_NODES * 64 + 255) / 256, 256, 0, stream>>>(
        node_feats, enW, enB, nodesH[0], N_NODES, D_NODE);
    gnn_embed<<<((long)N_EDGES * 64 + 255) / 256, 256, 0, stream>>>(
        edge_feats, eeW, eeB, edgesH[0], N_EDGES, D_EDGE);

    int cur = 0;
    for (int s = 0; s < STEPS; ++s) {
        int nxt = cur ^ 1;
        int base = 6 + 22 * s;
        // zero accumulators (sent|recv contiguous, nAgg|eAgg contiguous)
        gnn_zero_f32<<<2048, 256, 0, stream>>>(sent, (long)2 * N_NODES * 64);
        gnn_zero_f32<<<64, 256, 0, stream>>>(nAgg, (long)2 * N_GRAPHS * 64);

        // edge update: gather + MLP + scatter(sent/recv/eAgg)
        gnn_mlp<0><<<(N_EDGES + 63) / 64, 256, 0, stream>>>(
            edgesH[cur], nodesH[cur], nullptr, nullptr, gBuf[cur],
            senders, receivers, edge_gid,
            eW1p[s], (const float*)d_in[base + 1],
            (const float*)d_in[base + 2], (const float*)d_in[base + 3],
            eW2p[s], (const float*)d_in[base + 5],
            edgesH[nxt], sent, recv, eAgg, N_EDGES);

        // node update: gather + MLP + scatter(nAgg)
        gnn_mlp<1><<<(N_NODES + 63) / 64, 256, 0, stream>>>(
            nodesH[cur], nullptr, sent, recv, gBuf[cur],
            senders, receivers, node_gid,
            nW1p[s], (const float*)d_in[base + 7],
            (const float*)d_in[base + 8], (const float*)d_in[base + 9],
            nW2p[s], (const float*)d_in[base + 11],
            nodesH[nxt], nAgg, nullptr, nullptr, N_NODES);

        // global update
        float* gOut = (s == STEPS - 1) ? (float*)d_out : gBuf[nxt];
        gnn_global_mlp<<<N_GRAPHS, 128, 0, stream>>>(
            nAgg, eAgg, gBuf[cur],
            (const float*)d_in[base + 12], (const float*)d_in[base + 13],
            (const float*)d_in[base + 14], (const float*)d_in[base + 15],
            (const float*)d_in[base + 16], (const float*)d_in[base + 17],
            (const float*)d_in[base + 18], (const float*)d_in[base + 19],
            (const float*)d_in[base + 20], (const float*)d_in[base + 21],
            gOut);
        cur = nxt;
    }
}
